// ContrastiveLoss_75376676045036
// MI455X (gfx1250) — compile-verified, matrix operand order checked by reading, not run on hardware
//
#include <hip/hip_runtime.h>
#include <hip/hip_bf16.h>
#include <cstdint>

typedef __attribute__((ext_vector_type(16))) _Float16 v16h;
typedef __attribute__((ext_vector_type(8)))  float    v8f;

#define BN      8192
#define DK      1024
#define INV_T   10.0f       // 1 / temperature
#define TM      128         // rows per workgroup
#define TNB     128         // column block
#define KC      64          // K chunk staged in LDS
#define LPITCH  72          // LDS row pitch in halves (144 B, 16B-aligned, padded)
#define NCHUNK  8           // column chunks across gridDim.y
#define COLS_PER_CHUNK (BN / NCHUNK)
#define NKC     (DK / KC)   // 16 K-chunks

#if __has_builtin(__builtin_amdgcn_global_load_async_to_lds_b128)
#define HAVE_ASYNC_LDS 1
typedef int vi4 __attribute__((__vector_size__(4 * sizeof(int))));
typedef __attribute__((address_space(1))) vi4 as1_vi4;   // global (__device__)
typedef __attribute__((address_space(3))) vi4 as3_vi4;   // LDS (__shared__)
#else
#define HAVE_ASYNC_LDS 0
#endif

#if __has_builtin(__builtin_amdgcn_sched_barrier)
#define SCHED_FENCE() __builtin_amdgcn_sched_barrier(0)
#else
#define SCHED_FENCE()
#endif

__device__ __forceinline__ void wait_async_all() {
#if HAVE_ASYNC_LDS
#if __has_builtin(__builtin_amdgcn_s_wait_asynccnt)
    __builtin_amdgcn_s_wait_asynccnt(0);
#else
    asm volatile("s_wait_asynccnt 0" ::: "memory");
#endif
#endif
}

// ---------------- zero accumulators + output ----------------
__global__ void zero_kernel(float* se, float* ps, float* ct, float* out) {
    int i = blockIdx.x * 256 + threadIdx.x;
    if (i < BN) { se[i] = 0.f; ps[i] = 0.f; ct[i] = 0.f; }
    if (i == 0) out[0] = 0.f;
}

// ---------------- L2-normalize rows, f32 -> f16 ----------------
__global__ void normalize_kernel(const float* __restrict__ x,
                                 _Float16* __restrict__ xh) {
    __shared__ float red[256];
    const int row = blockIdx.x;
    const int t   = threadIdx.x;
    const float4 v = ((const float4*)(x + (size_t)row * DK))[t];  // 4 floats/thread
    red[t] = v.x * v.x + v.y * v.y + v.z * v.z + v.w * v.w;
    __syncthreads();
    for (int s = 128; s > 0; s >>= 1) {
        if (t < s) red[t] += red[t + s];
        __syncthreads();
    }
    const float sc = 1.0f / fmaxf(sqrtf(red[0]), 1e-12f);
    _Float16* o = xh + (size_t)row * DK + t * 4;
    o[0] = (_Float16)(v.x * sc);
    o[1] = (_Float16)(v.y * sc);
    o[2] = (_Float16)(v.z * sc);
    o[3] = (_Float16)(v.w * sc);
}

// ---- fused: S = X̂X̂ᵀ/T tile GEMM (WMMA f16) + masked softmax stats ----
__global__ void __launch_bounds__(256)
simloss_kernel(const _Float16* __restrict__ xh,
               const unsigned char* __restrict__ mask,
               float* __restrict__ sumExpArr,
               float* __restrict__ psumArr,
               float* __restrict__ cntArr) {
    __shared__ _Float16 Asm[2][TM][LPITCH];
    __shared__ _Float16 Bsm[2][TNB][LPITCH];

    const int tid     = threadIdx.x;
    const int lane    = tid & 31;
    const int wave    = tid >> 5;          // 8 waves, each owns 16 rows
    const int ln      = lane & 15;
    const int halfSel = lane >> 4;         // C layout: lanes 16-31 hold M=8..15
    const int mBase   = blockIdx.x * TM;
    const int chunk0  = blockIdx.y * COLS_PER_CHUNK;
    const int waveRow = mBase + wave * 16;

    // per-lane persistent stats for this lane's 8 rows
    float sumExpAcc[8], psumAcc[8], cntAcc[8];
#pragma unroll
    for (int r = 0; r < 8; ++r) { sumExpAcc[r] = 0.f; psumAcc[r] = 0.f; cntAcc[r] = 0.f; }

    // cooperative staging of one KC-wide chunk of A and B into LDS buffer `buf`
    auto stage = [&](int buf, int kc, int colBase) {
        for (int e = tid; e < 1024; e += 256) {
            const int row = e >> 3, seg = e & 7;
            const uint4* ga = (const uint4*)(xh + (size_t)(mBase + row) * DK + kc + seg * 8);
            const uint4* gb = (const uint4*)(xh + (size_t)(colBase + row) * DK + kc + seg * 8);
#if HAVE_ASYNC_LDS
            __builtin_amdgcn_global_load_async_to_lds_b128(
                (as1_vi4*)(uintptr_t)ga,
                (as3_vi4*)(uint32_t)(uintptr_t)&Asm[buf][row][seg * 8], 0, 0);
            __builtin_amdgcn_global_load_async_to_lds_b128(
                (as1_vi4*)(uintptr_t)gb,
                (as3_vi4*)(uint32_t)(uintptr_t)&Bsm[buf][row][seg * 8], 0, 0);
#else
            *(uint4*)(&Asm[buf][row][seg * 8]) = *ga;
            *(uint4*)(&Bsm[buf][row][seg * 8]) = *gb;
#endif
        }
    };

    for (int nb = 0; nb < COLS_PER_CHUNK; nb += TNB) {
        const int colBase = chunk0 + nb;
        v8f acc[8];
#pragma unroll
        for (int t = 0; t < 8; ++t)
#pragma unroll
            for (int i = 0; i < 8; ++i) acc[t][i] = 0.f;

        // prime the pipeline: chunk 0 -> buffer 0; warm the mask block into L2
        stage(0, 0, colBase);
        if (tid < 2 * TM) {
            const int pr = tid & (TM - 1), ph = (tid >> 7) * 64;
            __builtin_prefetch(&mask[(size_t)(mBase + pr) * BN + colBase + ph], 0, 1);
        }
        wait_async_all();
        __syncthreads();

        for (int i = 0; i < NKC; ++i) {
            const int cur = i & 1;
            if (i + 1 < NKC) stage(cur ^ 1, (i + 1) * KC, colBase);  // overlap next chunk

            // compute on buffer `cur`: 2 K-steps; per K-step load A + all 8 B
            // fragments (burst of 18 ds_load_b128), then 8 back-to-back WMMAs.
            // sched_barrier(0) keeps the loads from being sunk next to each WMMA,
            // so only one s_wait_dscnt per 8-WMMA burst is needed.
#pragma unroll
            for (int kk = 0; kk < KC; kk += 32) {
                union F { v16h v; uint4 u[2]; };
                F af, bf[8];
                const uint4* ap = (const uint4*)(&Asm[cur][wave * 16 + ln][kk + halfSel * 16]);
                af.u[0] = ap[0]; af.u[1] = ap[1];
#pragma unroll
                for (int t = 0; t < 8; ++t) {
                    const uint4* bp =
                        (const uint4*)(&Bsm[cur][t * 16 + ln][kk + halfSel * 16]);
                    bf[t].u[0] = bp[0]; bf[t].u[1] = bp[1];
                }
                SCHED_FENCE();   // all fragment loads scheduled before any WMMA
#pragma unroll
                for (int t = 0; t < 8; ++t)
                    acc[t] = __builtin_amdgcn_wmma_f32_16x16x32_f16(
                        false, af.v, false, bf[t].v, (short)0, acc[t], false, false);
            }
            wait_async_all();   // next buffer's async stores to LDS complete
            __syncthreads();    // all waves done reading `cur`, next buffer ready
        }

        // epilogue: fixed-max (cos/T <= 10) softmax stats + positive mask
#pragma unroll
        for (int t = 0; t < 8; ++t) {
            const int n = colBase + t * 16 + ln;
#pragma unroll
            for (int r = 0; r < 8; ++r) {
                const int m = waveRow + halfSel * 8 + r;
                const float v = acc[t][r] * INV_T;
                if (n != m) {                        // diagonal excluded
                    sumExpAcc[r] += __expf(v - INV_T);
                    if (mask[(size_t)m * BN + n]) {
                        psumAcc[r] += v;
                        cntAcc[r]  += 1.0f;
                    }
                }
            }
        }
    }

    // butterfly reduce across the 16 lanes sharing each row set, then flush
#pragma unroll
    for (int r = 0; r < 8; ++r) {
        float se = sumExpAcc[r], ps = psumAcc[r], ct = cntAcc[r];
        for (int off = 1; off < 16; off <<= 1) {
            se += __shfl_xor(se, off, 32);
            ps += __shfl_xor(ps, off, 32);
            ct += __shfl_xor(ct, off, 32);
        }
        if (ln == 0) {
            const int m = waveRow + halfSel * 8 + r;
            atomicAdd(&sumExpArr[m], se);
            atomicAdd(&psumArr[m],  ps);
            atomicAdd(&cntArr[m],   ct);
        }
    }
}

// ---------------- per-row loss -> scalar ----------------
__global__ void finalize_kernel(const float* __restrict__ se,
                                const float* __restrict__ ps,
                                const float* __restrict__ ct,
                                float* __restrict__ out) {
    __shared__ float red[256];
    const int i = blockIdx.x * 256 + threadIdx.x;
    // logZ = max(10) + log(sum exp(s - 10)); rows with no positives give 0
    const float l = ct[i] * (INV_T + __logf(se[i])) - ps[i];
    red[threadIdx.x] = l;
    __syncthreads();
    for (int s = 128; s > 0; s >>= 1) {
        if (threadIdx.x < s) red[threadIdx.x] += red[threadIdx.x + s];
        __syncthreads();
    }
    if (threadIdx.x == 0) atomicAdd(out, red[0] / (float)BN);
}

extern "C" void kernel_launch(void* const* d_in, const int* in_sizes, int n_in,
                              void* d_out, int out_size, void* d_ws, size_t ws_size,
                              hipStream_t stream) {
    const float*         emb  = (const float*)d_in[0];
    const unsigned char* mask = (const unsigned char*)d_in[1];   // jax bool -> 1 byte
    float* out = (float*)d_out;

    _Float16* xh = (_Float16*)d_ws;                               // 16 MB normalized f16
    float* se = (float*)((char*)d_ws + (size_t)BN * DK * sizeof(_Float16));
    float* ps = se + BN;
    float* ct = ps + BN;

    zero_kernel<<<(BN + 255) / 256, 256, 0, stream>>>(se, ps, ct, out);
    normalize_kernel<<<BN, 256, 0, stream>>>(emb, xh);
    dim3 grid(BN / TM, NCHUNK);
    simloss_kernel<<<grid, 256, 0, stream>>>(xh, mask, se, ps, ct);
    finalize_kernel<<<BN / 256, 256, 0, stream>>>(se, ps, ct, out);
}
